// Rotanet_73684458930525
// MI455X (gfx1250) — compile-verified, matrix operand order checked by reading, not run on hardware
//
#include <hip/hip_runtime.h>
#include <math.h>

typedef __attribute__((ext_vector_type(16))) _Float16 v16h;
typedef __attribute__((ext_vector_type(8)))  _Float16 v8h;
typedef __attribute__((ext_vector_type(8)))  float    v8f;

#define HID    96
#define NODES  81
#define NPAD   96
#define DEG    20
#define NEDGE  (NODES * DEG)          // 1620
#define ETILES ((NEDGE + 15) / 16)    // 102
#define WAVES  8

// ---------------- LDS layout (bytes) ----------------
#define OFF_H     0        // 96x96 f16   : h state
#define OFF_X     18432    // 96x96 f16   : X (input-MLP output, constant per run)
#define OFF_C     36864    // 96x96 f32   : LSTM cell state
#define OFF_S     73728    // 96x96 f16   : self part of rel layer0 (and temp)
#define OFF_N     92160    // 96x96 f16   : nbr part of rel layer0 (and temp)
#define OFF_T0    110592   // 96x96 f16   : temp
#define OFF_T1    129024   // 96x96 f16   : temp (M)
#define OFF_G     147456   // 96x96 f32   : edge aggregate
#define OFF_GATES 184320   // 96x384 f16  : LSTM gates
#define OFF_NBR   258048   // 81x20 i16 (padded to 3328B)
#define SMEM_BYTES 261376

// ---------------- WMMA fragment helpers ----------------
__device__ __forceinline__ v8f wmma_(v16h a, v16h b, v8f c) {
  return __builtin_amdgcn_wmma_f32_16x16x32_f16(false, a, false, b, (short)0, c,
                                                false, false);
}

// A fragment (16x32, f16 source in LDS): lane 0-15 -> M=lane, K=0..7 & 16..23;
// lane 16-31 -> same rows, K=8..15 & 24..31.
__device__ __forceinline__ v16h load_a_f16(const _Float16* A, int lda, int r0, int k0) {
  const int lane = threadIdx.x & 31;
  const _Float16* p = A + (r0 + (lane & 15)) * lda + k0 + ((lane >> 4) << 3);
  union { v16h v; v8h h[2]; } u;
  u.h[0] = *(const v8h*)(p);
  u.h[1] = *(const v8h*)(p + 16);
  return u.v;
}

__device__ __forceinline__ v16h load_a_f32(const float* A, int lda, int r0, int k0) {
  const int lane = threadIdx.x & 31;
  const float* p = A + (r0 + (lane & 15)) * lda + k0 + ((lane >> 4) << 3);
  v16h a;
#pragma unroll
  for (int j = 0; j < 8; ++j) {
    a[j]     = (_Float16)p[j];
    a[8 + j] = (_Float16)p[16 + j];
  }
  return a;
}

// B fragment from pre-swizzled global buffer:
// frag[((kt*ntB + nt)*32 + lane)*16 + j]  -> two contiguous 16B loads per lane.
__device__ __forceinline__ v16h load_b(const _Float16* B, int kt, int nt, int ntB) {
  const _Float16* p = B + ((((kt * ntB + nt) << 5) + (threadIdx.x & 31)) << 4);
  union { v16h v; v8h h[2]; } u;
  u.h[0] = *(const v8h*)(p);
  u.h[1] = *(const v8h*)(p + 8);
  return u.v;
}

struct GemmSrc {
  const _Float16* A; int lda;
  const _Float16* B; int ntB; int kt0;
};

// Generic workgroup GEMM: D(f16 LDS) = act( A1@B1 [+ A2@B2] + bias ).
// Each wave owns output-tile rows (fixed mt), reusing A fragments across all nt.
template <int KT1, int KT2>
__device__ void gemm1(GemmSrc s1, GemmSrc s2,
                      const float* bias1, const float* bias2, float bscale, bool do_relu,
                      _Float16* Dst, int ldd, int mtiles, int ntiles) {
  const int wave = threadIdx.x >> 5, lane = threadIdx.x & 31;
  for (int mt = wave; mt < mtiles; mt += WAVES) {
    v16h a1[KT1];
    v16h a2[KT2 > 0 ? KT2 : 1];
#pragma unroll
    for (int kt = 0; kt < KT1; ++kt) a1[kt] = load_a_f16(s1.A, s1.lda, mt * 16, kt * 32);
    if constexpr (KT2 > 0) {
#pragma unroll
      for (int kt = 0; kt < KT2; ++kt) a2[kt] = load_a_f16(s2.A, s2.lda, mt * 16, kt * 32);
    }
    const int rowbase = mt * 16 + ((lane >> 4) << 3);
    for (int nt = 0; nt < ntiles; ++nt) {
      v8f acc = {};
#pragma unroll
      for (int kt = 0; kt < KT1; ++kt)
        acc = wmma_(a1[kt], load_b(s1.B, s1.kt0 + kt, nt, s1.ntB), acc);
      if constexpr (KT2 > 0) {
#pragma unroll
        for (int kt = 0; kt < KT2; ++kt)
          acc = wmma_(a2[kt], load_b(s2.B, s2.kt0 + kt, nt, s2.ntB), acc);
      }
      const int col = nt * 16 + (lane & 15);
      float bb = 0.f;
      if (bias1) bb += bscale * bias1[col];
      if (bias2) bb += bias2[col];
#pragma unroll
      for (int r = 0; r < 8; ++r) {
        float v = acc[r] + bb;
        if (do_relu) v = fmaxf(v, 0.f);
        Dst[(rowbase + r) * ldd + col] = (_Float16)v;
      }
    }
  }
}

// M = Gagg(f32) @ W2 + 20*b2  -> f16
__device__ void gemm_af32(const float* A, const _Float16* B, const float* bias,
                          float bscale, _Float16* Dst) {
  const int wave = threadIdx.x >> 5, lane = threadIdx.x & 31;
  for (int mt = wave; mt < 6; mt += WAVES) {
    v16h a[3];
#pragma unroll
    for (int kt = 0; kt < 3; ++kt) a[kt] = load_a_f32(A, 96, mt * 16, kt * 32);
    const int rowbase = mt * 16 + ((lane >> 4) << 3);
#pragma unroll
    for (int nt = 0; nt < 6; ++nt) {
      v8f acc = {};
#pragma unroll
      for (int kt = 0; kt < 3; ++kt)
        acc = wmma_(a[kt], load_b(B, kt, nt, 6), acc);
      const int col = nt * 16 + (lane & 15);
      const float bb = bscale * bias[col];
#pragma unroll
      for (int r = 0; r < 8; ++r)
        Dst[(rowbase + r) * 96 + col] = (_Float16)(acc[r] + bb);
    }
  }
}

// Per-edge GEMM: h1 = relu( relu(S[n]+Nb[m]) @ W1 + b1 ), accumulated per-node.
__device__ void edge_phase(const _Float16* Ss, const _Float16* Ns, float* Gs,
                           const short* NB, const _Float16* W, const float* b1) {
  const int wave = threadIdx.x >> 5, lane = threadIdx.x & 31;
  for (int t = wave; t < ETILES; t += WAVES) {
    const int base = t * 16;
    int e = base + (lane & 15);
    int node = 0, nb = 0;
    if (e < NEDGE) { node = e / DEG; nb = (int)NB[e]; }
    const int kh = (lane >> 4) << 3;
    // Per-output-row node ids (shared across all 6 N-tiles).
    int  rowNode[8];
    bool rowOk[8];
    const int mbase = base + ((lane >> 4) << 3);
#pragma unroll
    for (int r = 0; r < 8; ++r) {
      const int e2 = mbase + r;
      rowOk[r] = e2 < NEDGE;
      rowNode[r] = rowOk[r] ? (e2 / DEG) : 0;
    }
    v8f acc[6] = {};
#pragma unroll
    for (int kt = 0; kt < 3; ++kt) {
      const int k0 = kt * 32 + kh;
      const v8h s0 = *(const v8h*)(Ss + node * 96 + k0);
      const v8h s1 = *(const v8h*)(Ss + node * 96 + k0 + 16);
      const v8h n0 = *(const v8h*)(Ns + nb * 96 + k0);
      const v8h n1 = *(const v8h*)(Ns + nb * 96 + k0 + 16);
      const v8h z = {};
      union { v16h v; v8h h[2]; } u;
      u.h[0] = __builtin_elementwise_max(s0 + n0, z);   // packed f16 relu
      u.h[1] = __builtin_elementwise_max(s1 + n1, z);
#pragma unroll
      for (int nt = 0; nt < 6; ++nt)
        acc[nt] = wmma_(u.v, load_b(W, kt, nt, 6), acc[nt]);
    }
#pragma unroll
    for (int nt = 0; nt < 6; ++nt) {
      const int col = nt * 16 + (lane & 15);
      const float bb = b1[col];
#pragma unroll
      for (int r = 0; r < 8; ++r) {
        if (rowOk[r])
          atomicAdd(Gs + rowNode[r] * 96 + col, fmaxf(acc[nt][r] + bb, 0.f));
      }
    }
  }
}

// ---------------- weight pre-swizzle ----------------
__global__ void Rotanet_73684458930525_prep(const float* __restrict__ src, int K, int Nn,
                                            int ktiles, int ntiles,
                                            _Float16* __restrict__ dst) {
  const int idx = blockIdx.x * blockDim.x + threadIdx.x;
  const int total = ktiles * ntiles * 512;
  if (idx >= total) return;
  const int j = idx & 15;
  const int lane = (idx >> 4) & 31;
  const int tile = idx >> 9;
  const int nt = tile % ntiles, kt = tile / ntiles;
  const int kof = (j & 7) + ((lane >> 4) << 3) + ((j >> 3) << 4);
  const int k = kt * 32 + kof;
  const int n = nt * 16 + (lane & 15);
  const float v = (k < K && n < Nn) ? src[k * Nn + n] : 0.f;
  dst[idx] = (_Float16)v;
}

// ---------------- main fused kernel: one WG per batch element ----------------
struct Params {
  const int* grids; const int* iters; const float* emb;
  const float *in_b0, *in_b1, *in_b2;
  const float *rel_b0, *rel_b1, *rel_b2;
  const float *g_b0, *g_b1, *g_b2;
  const float *lstm_bih, *lstm_bhh;
  const float *r_b0, *r_b1, *r_b2;
  const float* c0;
  const _Float16 *w_in0, *w_in1, *w_in2;
  const _Float16 *w_rel0, *w_rel1, *w_rel2;
  const _Float16 *w_g0, *w_g1, *w_g2;
  const _Float16 *w_wih, *w_whh;
  const _Float16 *w_r0, *w_r1, *w_r2;
  float* out; int B;
};

__global__ __launch_bounds__(256, 1)
void Rotanet_73684458930525_solve(Params p) {
  extern __shared__ char smem[];
  _Float16* Hs = (_Float16*)(smem + OFF_H);
  _Float16* Xs = (_Float16*)(smem + OFF_X);
  float*    Cs = (float*)(smem + OFF_C);
  _Float16* Ss = (_Float16*)(smem + OFF_S);
  _Float16* Ns = (_Float16*)(smem + OFF_N);
  _Float16* T0 = (_Float16*)(smem + OFF_T0);
  _Float16* T1 = (_Float16*)(smem + OFF_T1);
  float*    Gs = (float*)(smem + OFF_G);
  _Float16* GT = (_Float16*)(smem + OFF_GATES);
  short*    NB = (short*)(smem + OFF_NBR);

  const int tid = threadIdx.x;
  const int b   = blockIdx.x;
  const GemmSrc none = {};

  // Build sudoku neighbor table (20 per node; order irrelevant to the sum).
  if (tid < NODES) {
    const int r = tid / 9, c = tid % 9;
    int cnt = 0;
    for (int k = 0; k < 9; ++k) if (k != c) NB[tid * DEG + cnt++] = (short)(r * 9 + k);
    for (int k = 0; k < 9; ++k) if (k != r) NB[tid * DEG + cnt++] = (short)(k * 9 + c);
    const int br = (r / 3) * 3, bc = (c / 3) * 3;
    for (int dr = 0; dr < 3; ++dr)
      for (int dc = 0; dc < 3; ++dc) {
        const int rr = br + dr, cc = bc + dc;
        if (rr != r && cc != c) NB[tid * DEG + cnt++] = (short)(rr * 9 + cc);
      }
  }
  // Load c0, build embedding A matrix (96x32, K padded 16->32) in T0.
  for (int i = tid; i < NPAD * HID; i += 256) {
    const int n = i / HID, j = i % HID;
    Cs[i] = (n < NODES) ? p.c0[((size_t)b * NODES + n) * HID + j] : 0.f;
  }
  for (int i = tid; i < NPAD * 32; i += 256) {
    const int n = i >> 5, k = i & 31;
    float v = 0.f;
    if (n < NODES && k < 16) v = p.emb[p.grids[b * NODES + n] * 16 + k];
    T0[i] = (_Float16)v;
  }
  __syncthreads();

  // X = mlp(emb[grids]) ; h0 = X
  gemm1<1, 0>(GemmSrc{T0, 32, p.w_in0, 6, 0}, none, p.in_b0, nullptr, 1.f, true,  T1, 96, 6, 6);
  __syncthreads();
  gemm1<3, 0>(GemmSrc{T1, 96, p.w_in1, 6, 0}, none, p.in_b1, nullptr, 1.f, true,  Ss, 96, 6, 6);
  __syncthreads();
  gemm1<3, 0>(GemmSrc{Ss, 96, p.w_in2, 6, 0}, none, p.in_b2, nullptr, 1.f, false, Xs, 96, 6, 6);
  __syncthreads();
  for (int i = tid; i < NPAD * HID; i += 256) Hs[i] = Xs[i];
  __syncthreads();

  const int iters = p.iters[0];
  for (int it = 0; it < iters; ++it) {
    // rel layer0 factorized: S = h@W0_top + b0 ; Nb = h@W0_bot
    gemm1<3, 0>(GemmSrc{Hs, 96, p.w_rel0, 6, 0}, none, p.rel_b0, nullptr, 1.f, false, Ss, 96, 6, 6);
    gemm1<3, 0>(GemmSrc{Hs, 96, p.w_rel0, 6, 3}, none, nullptr,  nullptr, 1.f, false, Ns, 96, 6, 6);
    for (int i = tid; i < NPAD * HID; i += 256) Gs[i] = 0.f;
    __syncthreads();

    // per-edge layer1 + per-node aggregation of relu(h1)
    edge_phase(Ss, Ns, Gs, NB, p.w_rel1, p.rel_b1);
    __syncthreads();

    // rel layer2 post-aggregation: M = Gagg@W2 + 20*b2
    gemm_af32(Gs, p.w_rel2, p.rel_b2, 20.f, T1);
    __syncthreads();

    // g MLP: x_in = mlp(concat(X, M))
    gemm1<3, 3>(GemmSrc{Xs, 96, p.w_g0, 6, 0}, GemmSrc{T1, 96, p.w_g0, 6, 3},
                p.g_b0, nullptr, 1.f, true, T0, 96, 6, 6);
    __syncthreads();
    gemm1<3, 0>(GemmSrc{T0, 96, p.w_g1, 6, 0}, none, p.g_b1, nullptr, 1.f, true,  Ss, 96, 6, 6);
    __syncthreads();
    gemm1<3, 0>(GemmSrc{Ss, 96, p.w_g2, 6, 0}, none, p.g_b2, nullptr, 1.f, false, Ns, 96, 6, 6);
    __syncthreads();

    // LSTM gates = x_in@Wih + h@Whh + (bih+bhh)
    gemm1<3, 3>(GemmSrc{Ns, 96, p.w_wih, 24, 0}, GemmSrc{Hs, 96, p.w_whh, 24, 0},
                p.lstm_bih, p.lstm_bhh, 1.f, false, GT, 384, 6, 24);
    __syncthreads();

    // LSTM cell elementwise
    for (int i = tid; i < NODES * HID; i += 256) {
      const int n = i / HID, j = i % HID;
      const _Float16* g = GT + n * 384;
      const float gi = (float)g[j], gf = (float)g[96 + j];
      const float gg = (float)g[192 + j], go = (float)g[288 + j];
      const float si = 1.f / (1.f + __expf(-gi));
      const float sf = 1.f / (1.f + __expf(-gf));
      const float so = 1.f / (1.f + __expf(-go));
      const float c2 = sf * Cs[i] + si * tanhf(gg);
      Cs[i] = c2;
      Hs[i] = (_Float16)(so * tanhf(c2));
    }
    __syncthreads();

    // readout MLP
    gemm1<3, 0>(GemmSrc{Hs, 96, p.w_r0, 6, 0}, none, p.r_b0, nullptr, 1.f, true, T0, 96, 6, 6);
    __syncthreads();
    gemm1<3, 0>(GemmSrc{T0, 96, p.w_r1, 6, 0}, none, p.r_b1, nullptr, 1.f, true, Ss, 96, 6, 6);
    __syncthreads();
    {
      const int wave = tid >> 5, lane = tid & 31;
      for (int mt = wave; mt < 6; mt += WAVES) {
        v8f acc = {};
#pragma unroll
        for (int kt = 0; kt < 3; ++kt)
          acc = wmma_(load_a_f16(Ss, 96, mt * 16, kt * 32), load_b(p.w_r2, kt, 0, 1), acc);
        const int col = lane & 15;
        if (col < 9) {
          const float bo = p.r_b2[col];
          const int rowbase = mt * 16 + ((lane >> 4) << 3);
#pragma unroll
          for (int r = 0; r < 8; ++r) {
            const int node = rowbase + r;
            if (node < NODES)
              p.out[(((size_t)it * p.B + b) * NODES + node) * 9 + col] = acc[r] + bo;
          }
        }
      }
    }
    __syncthreads();
  }
}

// ---------------- host ----------------
extern "C" void kernel_launch(void* const* d_in, const int* in_sizes, int n_in,
                              void* d_out, int out_size, void* d_ws, size_t ws_size,
                              hipStream_t stream) {
  (void)n_in; (void)out_size; (void)ws_size;
  _Float16* ws = (_Float16*)d_ws;

  struct WDesc { int src, K, N, kt, nt; size_t off; };
  static const WDesc wt[14] = {
      {3,  16,  96, 1, 6,  0},        // in_w0
      {5,  96,  96, 3, 6,  3072},     // in_w1
      {7,  96,  96, 3, 6,  12288},    // in_w2
      {9,  192, 96, 6, 6,  21504},    // rel_w0
      {11, 96,  96, 3, 6,  39936},    // rel_w1
      {13, 96,  96, 3, 6,  49152},    // rel_w2
      {15, 192, 96, 6, 6,  58368},    // g_w0
      {17, 96,  96, 3, 6,  76800},    // g_w1
      {19, 96,  96, 3, 6,  86016},    // g_w2
      {21, 96, 384, 3, 24, 95232},    // lstm_wih
      {22, 96, 384, 3, 24, 132096},   // lstm_whh
      {25, 96,  96, 3, 6,  168960},   // r_w0
      {27, 96,  96, 3, 6,  178176},   // r_w1
      {29, 96,   9, 3, 1,  187392},   // r_w2
  };
  for (int i = 0; i < 14; ++i) {
    const int total = wt[i].kt * wt[i].nt * 512;
    Rotanet_73684458930525_prep<<<(total + 255) / 256, 256, 0, stream>>>(
        (const float*)d_in[wt[i].src], wt[i].K, wt[i].N, wt[i].kt, wt[i].nt,
        ws + wt[i].off);
  }

  Params p;
  p.grids = (const int*)d_in[0];
  p.iters = (const int*)d_in[1];
  p.emb = (const float*)d_in[2];
  p.in_b0 = (const float*)d_in[4];  p.in_b1 = (const float*)d_in[6];  p.in_b2 = (const float*)d_in[8];
  p.rel_b0 = (const float*)d_in[10]; p.rel_b1 = (const float*)d_in[12]; p.rel_b2 = (const float*)d_in[14];
  p.g_b0 = (const float*)d_in[16]; p.g_b1 = (const float*)d_in[18]; p.g_b2 = (const float*)d_in[20];
  p.lstm_bih = (const float*)d_in[23]; p.lstm_bhh = (const float*)d_in[24];
  p.r_b0 = (const float*)d_in[26]; p.r_b1 = (const float*)d_in[28]; p.r_b2 = (const float*)d_in[30];
  p.c0 = (const float*)d_in[31];
  p.w_in0 = ws + 0;      p.w_in1 = ws + 3072;   p.w_in2 = ws + 12288;
  p.w_rel0 = ws + 21504; p.w_rel1 = ws + 39936; p.w_rel2 = ws + 49152;
  p.w_g0 = ws + 58368;   p.w_g1 = ws + 76800;   p.w_g2 = ws + 86016;
  p.w_wih = ws + 95232;  p.w_whh = ws + 132096;
  p.w_r0 = ws + 168960;  p.w_r1 = ws + 178176;  p.w_r2 = ws + 187392;
  p.out = (float*)d_out;
  p.B = in_sizes[0] / NODES;

  hipFuncSetAttribute((const void*)Rotanet_73684458930525_solve,
                      hipFuncAttributeMaxDynamicSharedMemorySize, SMEM_BYTES);
  Rotanet_73684458930525_solve<<<p.B, 256, SMEM_BYTES, stream>>>(p);
}
